// ResidualVQ_12902081757241
// MI455X (gfx1250) — matrix operand order repaired and checked by reading, not compiled, not run on hardware
//
// ResidualVQ forward for MI455X (gfx1250, wave32, WMMA).
// Strategy:
//   - All GEMM work via v_wmma_f32_16x16x32_bf16 (bf16 inputs, f32 accum).
//   - Phase 1: weight-norm prep (bf16 W_in/W_out), codebook bf16 copy + |c|^2,
//              x transposed to [B,T,CIN] bf16 so A-fragments are contiguous.
//   - Phase 2: 32 fused kernels: residual tile converted bf16->LDS once,
//              distance-GEMM (scores = 0.5|c|^2 - r.c) with A-frags from LDS
//              (ds_load_b128) and codebook B-frags streamed from L2,
//              per-row argmin (shuffle reduce) + fp32 gather/residual update.
//              64-row M-tile halves codebook L2 traffic vs 32-row.
//   - Phase 3: zq = res0 - res (bf16), then output GEMM + bias.
#include <hip/hip_runtime.h>
#include <hip/hip_bf16.h>
#include <stdint.h>

#define B_    16
#define CIN_  1280
#define T_    1024
#define D_    256
#define Q_    32
#define K_    1024
#define COUT_ 1280

#define ATILE_LD 264   // 256 + 8 halfs pad: row stride 132 dwords -> bank step 4

typedef __bf16 v16bf __attribute__((ext_vector_type(16)));
typedef float  v8f   __attribute__((ext_vector_type(8)));

union FragBF {
    uint4 q[2];
    unsigned short u[16];
    v16bf v;
};

__device__ __forceinline__ unsigned short f2bf(float f) {
    union { float f; unsigned u; } x; x.f = f;
    unsigned u = x.u;
    unsigned r = u + 0x7FFFu + ((u >> 16) & 1u);   // round-to-nearest-even
    return (unsigned short)(r >> 16);
}

__device__ __forceinline__ v8f wmma_bf16(v16bf a, v16bf b, v8f c) {
    return __builtin_amdgcn_wmma_f32_16x16x32_bf16(
        /*neg_a=*/false, a, /*neg_b=*/false, b,
        /*c_mod=*/(short)0, c, /*reuse_a=*/false, /*reuse_b=*/false);
}

// A-operand fragment: row-major [M,K] bf16; lane holds row (lane&15),
// K chunks [koff, koff+8) and [koff+16, koff+24), koff = (lane>>4)*8.
__device__ __forceinline__ v16bf load_a_bf16(const unsigned short* base, int row,
                                             int ld, int k0, int lane) {
    FragBF f;
    const unsigned short* p = base + (size_t)row * ld + k0 + ((lane >> 4) << 3);
    f.q[0] = *(const uint4*)(p);
    f.q[1] = *(const uint4*)(p + 16);
    return f.v;
}

// Same A-fragment pattern, but from an LDS tile with padded row stride.
__device__ __forceinline__ v16bf load_a_lds(const unsigned short (*tile)[ATILE_LD],
                                            int row, int k0, int lane) {
    FragBF f;
    const unsigned short* p = &tile[row][k0 + ((lane >> 4) << 3)];
    f.q[0] = *(const uint4*)(p);
    f.q[1] = *(const uint4*)(p + 16);
    return f.v;
}

// B-operand fragment: B^T row-major [N,K] bf16; lane holds column (lane&15),
// K chunks [koff, koff+8) and [koff+16, koff+24), koff = (lane>>4)*8.
__device__ __forceinline__ v16bf load_b_bf16(const unsigned short* base, int row,
                                             int ld, int k0, int lane) {
    FragBF f;
    const unsigned short* p = base + (size_t)row * ld + k0 + ((lane >> 4) << 3);
    f.q[0] = *(const uint4*)(p);
    f.q[1] = *(const uint4*)(p + 16);
    return f.v;
}

// ---------------- prep kernels ----------------

// One block per output row: w[row,:] = (g[row]/||v[row,:]||) * v[row,:]  (bf16)
__global__ void prep_weight_kernel(const float* __restrict__ v,
                                   const float* __restrict__ g,
                                   unsigned short* __restrict__ w, int cols) {
    __shared__ float sred[256];
    int row = blockIdx.x;
    const float* vr = v + (size_t)row * cols;
    float s = 0.f;
    for (int c = threadIdx.x; c < cols; c += 256) { float t = vr[c]; s += t * t; }
    sred[threadIdx.x] = s;
    __syncthreads();
    for (int off = 128; off > 0; off >>= 1) {
        if (threadIdx.x < off) sred[threadIdx.x] += sred[threadIdx.x + off];
        __syncthreads();
    }
    float scale = g[row] / sqrtf(sred[0]);
    for (int c = threadIdx.x; c < cols; c += 256)
        w[(size_t)row * cols + c] = f2bf(vr[c] * scale);
}

// One block per codebook row: bf16 copy + squared norm.
__global__ void prep_cb_kernel(const float* __restrict__ cb,
                               unsigned short* __restrict__ cbb,
                               float* __restrict__ cbn) {
    __shared__ float sred[256];
    int row = blockIdx.x;
    float t = cb[(size_t)row * D_ + threadIdx.x];
    cbb[(size_t)row * D_ + threadIdx.x] = f2bf(t);
    sred[threadIdx.x] = t * t;
    __syncthreads();
    for (int off = 128; off > 0; off >>= 1) {
        if (threadIdx.x < off) sred[threadIdx.x] += sred[threadIdx.x + off];
        __syncthreads();
    }
    if (threadIdx.x == 0) cbn[row] = sred[0];
}

// x [B,CIN,T] f32 -> xT [B,T,CIN] bf16 (LDS-tiled transpose)
__global__ void transpose_x_kernel(const float* __restrict__ x,
                                   unsigned short* __restrict__ xT) {
    __shared__ unsigned short tile[32][33];
    int b = blockIdx.z, t0 = blockIdx.x * 32, c0 = blockIdx.y * 32;
    #pragma unroll
    for (int i = 0; i < 4; i++) {
        int c = c0 + threadIdx.y + i * 8;
        tile[threadIdx.y + i * 8][threadIdx.x] =
            f2bf(x[((size_t)b * CIN_ + c) * T_ + t0 + threadIdx.x]);
    }
    __syncthreads();
    #pragma unroll
    for (int i = 0; i < 4; i++) {
        int t = t0 + threadIdx.y + i * 8;
        xT[((size_t)b * T_ + t) * CIN_ + c0 + threadIdx.x] = tile[threadIdx.x][threadIdx.y + i * 8];
    }
}

// ---------------- GEMM kernels ----------------

// res0[b,t,d] = sum_c xT[b,t,c]*Win[d,c] + bias[d]; also res = res0.
// 128 threads = 4 waves in 2x2; each wave does 32x32 (2x2 fragments).
__global__ __launch_bounds__(128) void in_proj_kernel(
    const unsigned short* __restrict__ xT, const unsigned short* __restrict__ win,
    const float* __restrict__ bias, float* __restrict__ res0, float* __restrict__ res) {
    int b = blockIdx.z;
    int t_base = blockIdx.y * 64, d_base = blockIdx.x * 64;
    int tid = threadIdx.x, wave = tid >> 5, lane = tid & 31;
    int tm0 = t_base + (wave >> 1) * 32;
    int dn0 = d_base + (wave & 1) * 32;
    const unsigned short* xrow = xT + (size_t)b * T_ * CIN_;

    v8f acc[2][2] = {};
    for (int k0 = 0; k0 < CIN_; k0 += 32) {
        v16bf a0 = load_a_bf16(xrow, tm0 + (lane & 15), CIN_, k0, lane);
        v16bf a1 = load_a_bf16(xrow, tm0 + 16 + (lane & 15), CIN_, k0, lane);
        v16bf b0 = load_b_bf16(win, dn0 + (lane & 15), CIN_, k0, lane);
        v16bf b1 = load_b_bf16(win, dn0 + 16 + (lane & 15), CIN_, k0, lane);
        acc[0][0] = wmma_bf16(a0, b0, acc[0][0]);
        acc[0][1] = wmma_bf16(a0, b1, acc[0][1]);
        acc[1][0] = wmma_bf16(a1, b0, acc[1][0]);
        acc[1][1] = wmma_bf16(a1, b1, acc[1][1]);
    }
    #pragma unroll
    for (int mf = 0; mf < 2; mf++)
        #pragma unroll
        for (int nf = 0; nf < 2; nf++) {
            int dn = dn0 + nf * 16 + (lane & 15);
            float bv = bias[dn];
            #pragma unroll
            for (int r = 0; r < 8; r++) {
                int tm = tm0 + mf * 16 + r + ((lane >> 4) << 3);
                float val = acc[mf][nf][r] + bv;
                size_t o = ((size_t)b * T_ + tm) * D_ + dn;
                res0[o] = val;
                res[o] = val;
            }
        }
}

// One RVQ quantizer step, fused: scores + argmin + gather + residual update.
// 256 threads = 8 waves: wave>>2 selects the 32-row half of a 64-row M-tile,
// wave&3 selects a 256-entry codebook range (scanned as 4 blocks of 64).
// Residual tile is converted to bf16 in LDS once; A-frags come from LDS.
__global__ __launch_bounds__(256) void rvq_step_kernel(
    float* __restrict__ res, const unsigned short* __restrict__ cbb_q,
    const float* __restrict__ cbn_q, const float* __restrict__ cbf_q,
    int* __restrict__ codes_q) {
    __shared__ __align__(16) unsigned short atile[64][ATILE_LD];
    __shared__ float s_val[8][32];
    __shared__ int s_idx[8][32];
    __shared__ int s_fin[64];

    int bt0 = blockIdx.x * 64;
    int tid = threadIdx.x, wave = tid >> 5, lane = tid & 31;

    // stage residual tile (64 x 256 fp32) -> bf16 LDS, one pass
    {
        int row = tid >> 2;            // 0..63
        int c0 = (tid & 3) * 64;       // 4 threads per row
        const float* src = res + (size_t)(bt0 + row) * D_ + c0;
        unsigned short* dst = &atile[row][c0];
        #pragma unroll
        for (int i = 0; i < 16; i++) {
            float4 f = *(const float4*)(src + i * 4);
            dst[i * 4 + 0] = f2bf(f.x);
            dst[i * 4 + 1] = f2bf(f.y);
            dst[i * 4 + 2] = f2bf(f.z);
            dst[i * 4 + 3] = f2bf(f.w);
        }
    }
    __syncthreads();

    int row_base = (wave >> 2) * 32;   // which 32-row half this wave computes
    int nwave = wave & 3;              // which 256-entry codebook range

    float runv[2][8];
    int runi[2][8];
    #pragma unroll
    for (int mf = 0; mf < 2; mf++)
        #pragma unroll
        for (int r = 0; r < 8; r++) { runv[mf][r] = 3.4e38f; runi[mf][r] = 0; }

    for (int blk = 0; blk < 4; blk++) {
        int nbase = nwave * 256 + blk * 64;
        // hint: pull next codebook chunk toward the WGP
        __builtin_prefetch(cbb_q + ((size_t)((nbase + 64) & (K_ - 1))) * D_, 0, 1);
        v8f acc[2][4] = {};
        for (int k0 = 0; k0 < D_; k0 += 32) {
            v16bf a0 = load_a_lds(atile, row_base + (lane & 15), k0, lane);
            v16bf a1 = load_a_lds(atile, row_base + 16 + (lane & 15), k0, lane);
            #pragma unroll
            for (int j = 0; j < 4; j++) {
                v16bf bf = load_b_bf16(cbb_q, nbase + j * 16 + (lane & 15), D_, k0, lane);
                acc[0][j] = wmma_bf16(a0, bf, acc[0][j]);
                acc[1][j] = wmma_bf16(a1, bf, acc[1][j]);
            }
        }
        #pragma unroll
        for (int j = 0; j < 4; j++) {
            int n = nbase + j * 16 + (lane & 15);
            float cn = 0.5f * cbn_q[n];
            #pragma unroll
            for (int mf = 0; mf < 2; mf++)
                #pragma unroll
                for (int r = 0; r < 8; r++) {
                    float s = cn - acc[mf][j][r];
                    if (s < runv[mf][r] || (s == runv[mf][r] && n < runi[mf][r])) {
                        runv[mf][r] = s; runi[mf][r] = n;
                    }
                }
        }
    }
    // reduce across the 16 lanes sharing each C-row (xor within half-wave)
    #pragma unroll
    for (int mf = 0; mf < 2; mf++)
        #pragma unroll
        for (int r = 0; r < 8; r++) {
            float v = runv[mf][r];
            int i = runi[mf][r];
            #pragma unroll
            for (int m = 1; m <= 8; m <<= 1) {
                float ov = __shfl_xor(v, m, 32);
                int oi = __shfl_xor(i, m, 32);
                if (ov < v || (ov == v && oi < i)) { v = ov; i = oi; }
            }
            if ((lane & 15) == 0) {
                int mrow = mf * 16 + r + ((lane >> 4) << 3);   // row within half
                s_val[wave][mrow] = v;
                s_idx[wave][mrow] = i;
            }
        }
    __syncthreads();
    if (tid < 64) {  // cross-wave reduce: 4 waves per 32-row half, ascending n
        int half = tid >> 5, mrow = tid & 31;
        int w0 = half * 4;
        float bv = s_val[w0][mrow];
        int bi = s_idx[w0][mrow];
        #pragma unroll
        for (int w = 1; w < 4; w++) {
            float v = s_val[w0 + w][mrow];
            int i = s_idx[w0 + w][mrow];
            if (v < bv || (v == bv && i < bi)) { bv = v; bi = i; }
        }
        s_fin[tid] = bi;
        codes_q[bt0 + tid] = bi;  // codes_q is the [B*T] slab for this quantizer
    }
    __syncthreads();
    // fp32 residual update: res[row,:] -= codebook_f32[idx,:]
    // 256 threads = 2 rows per iteration, float2 per thread
    for (int m2 = 0; m2 < 32; m2++) {
        int row = m2 * 2 + (tid >> 7);
        int d2 = tid & 127;
        int idx = s_fin[row];
        const float2* src = (const float2*)(cbf_q + (size_t)idx * D_) + d2;
        float2* dst = (float2*)(res + (size_t)(bt0 + row) * D_) + d2;
        float2 a = *dst, c = *src;
        a.x -= c.x; a.y -= c.y;
        *dst = a;
    }
}

// zq_bf16 = bf16(res0 - res)  (forward value of the straight-through sum)
__global__ void zqb_kernel(const float* __restrict__ res0, const float* __restrict__ res,
                           unsigned short* __restrict__ zqb) {
    size_t i = ((size_t)blockIdx.x * 256 + threadIdx.x) * 4;
    float4 a = *(const float4*)(res0 + i);
    float4 b = *(const float4*)(res + i);
    ushort4 o;
    o.x = f2bf(a.x - b.x); o.y = f2bf(a.y - b.y);
    o.z = f2bf(a.z - b.z); o.w = f2bf(a.w - b.w);
    *(ushort4*)(zqb + i) = o;
}

// out[b,o,t] = sum_d Wout[o,d]*zq[b,t,d] + bias[o]
__global__ __launch_bounds__(128) void out_proj_kernel(
    const unsigned short* __restrict__ wout, const unsigned short* __restrict__ zqb,
    const float* __restrict__ bias, float* __restrict__ out) {
    int b = blockIdx.z;
    int o_base = blockIdx.y * 64, t_base = blockIdx.x * 64;
    int tid = threadIdx.x, wave = tid >> 5, lane = tid & 31;
    int om0 = o_base + (wave >> 1) * 32;
    int tn0 = t_base + (wave & 1) * 32;
    const unsigned short* zrow = zqb + (size_t)b * T_ * D_;

    v8f acc[2][2] = {};
    for (int k0 = 0; k0 < D_; k0 += 32) {
        v16bf a0 = load_a_bf16(wout, om0 + (lane & 15), D_, k0, lane);
        v16bf a1 = load_a_bf16(wout, om0 + 16 + (lane & 15), D_, k0, lane);
        v16bf b0 = load_b_bf16(zrow, tn0 + (lane & 15), D_, k0, lane);
        v16bf b1 = load_b_bf16(zrow, tn0 + 16 + (lane & 15), D_, k0, lane);
        acc[0][0] = wmma_bf16(a0, b0, acc[0][0]);
        acc[0][1] = wmma_bf16(a0, b1, acc[0][1]);
        acc[1][0] = wmma_bf16(a1, b0, acc[1][0]);
        acc[1][1] = wmma_bf16(a1, b1, acc[1][1]);
    }
    #pragma unroll
    for (int mf = 0; mf < 2; mf++)
        #pragma unroll
        for (int nf = 0; nf < 2; nf++) {
            int tn = tn0 + nf * 16 + (lane & 15);
            #pragma unroll
            for (int r = 0; r < 8; r++) {
                int om = om0 + mf * 16 + r + ((lane >> 4) << 3);
                out[((size_t)b * COUT_ + om) * T_ + tn] = acc[mf][nf][r] + bias[om];
            }
        }
}

extern "C" void kernel_launch(void* const* d_in, const int* in_sizes, int n_in,
                              void* d_out, int out_size, void* d_ws, size_t ws_size,
                              hipStream_t stream) {
    const float* x         = (const float*)d_in[0];
    const float* in_v      = (const float*)d_in[1];
    const float* in_g      = (const float*)d_in[2];
    const float* in_b      = (const float*)d_in[3];
    const float* out_v     = (const float*)d_in[4];
    const float* out_g     = (const float*)d_in[5];
    const float* out_b     = (const float*)d_in[6];
    const float* codebooks = (const float*)d_in[7];

    float* out   = (float*)d_out;
    int*   codes = (int*)(out + (size_t)B_ * COUT_ * T_);  // [Q,B,T] after out

    // workspace carve-up (zqb aliases xT: xT dead after in_proj)
    char* ws = (char*)d_ws;
    float* res0 = (float*)ws;                 ws += (size_t)B_ * T_ * D_ * 4;
    float* res  = (float*)ws;                 ws += (size_t)B_ * T_ * D_ * 4;
    unsigned short* xT  = (unsigned short*)ws; ws += (size_t)B_ * T_ * CIN_ * 2;
    unsigned short* win = (unsigned short*)ws; ws += (size_t)D_ * CIN_ * 2;
    unsigned short* wout = (unsigned short*)ws; ws += (size_t)COUT_ * D_ * 2;
    unsigned short* cbb = (unsigned short*)ws; ws += (size_t)Q_ * K_ * D_ * 2;
    float* cbn = (float*)ws;
    unsigned short* zqb = xT;                  // reuse

    prep_weight_kernel<<<D_, 256, 0, stream>>>(in_v, in_g, win, CIN_);
    prep_weight_kernel<<<COUT_, 256, 0, stream>>>(out_v, out_g, wout, D_);
    prep_cb_kernel<<<Q_ * K_, 256, 0, stream>>>(codebooks, cbb, cbn);
    transpose_x_kernel<<<dim3(T_ / 32, CIN_ / 32, B_), dim3(32, 8), 0, stream>>>(x, xT);

    in_proj_kernel<<<dim3(D_ / 64, T_ / 64, B_), 128, 0, stream>>>(xT, win, in_b, res0, res);

    for (int q = 0; q < Q_; q++) {
        rvq_step_kernel<<<(B_ * T_) / 64, 256, 0, stream>>>(
            res, cbb + (size_t)q * K_ * D_, cbn + (size_t)q * K_,
            codebooks + (size_t)q * K_ * D_, codes + (size_t)q * B_ * T_);
    }

    zqb_kernel<<<((size_t)B_ * T_ * D_) / (256 * 4), 256, 0, stream>>>(res0, res, zqb);
    out_proj_kernel<<<dim3(T_ / 64, COUT_ / 64, B_), 128, 0, stream>>>(wout, zqb, out_b, out);
}
